// MLPTexture3D_76424648065133
// MI455X (gfx1250) — matrix operand order repaired and checked by reading, not compiled, not run on hardware
//
#include <hip/hip_runtime.h>
#include <hip/hip_bf16.h>
#include <cstdint>
#include <cstddef>

// ---------------- Types for CDNA5 WMMA ----------------
typedef __attribute__((ext_vector_type(16))) _Float16 v16h;
typedef __attribute__((ext_vector_type(8)))  _Float16 v8h;
typedef __attribute__((ext_vector_type(8)))  float    v8f;

// ---------------- Problem constants (match reference) ----------------
#define NUM_LEVELS       16
#define LOG2_HASHMAP     19
#define HASHMAP_SIZE     (1u << LOG2_HASHMAP)   // 524288
#define LEVEL_DIM        2
#define HID              32
#define CHANNELS         9
#define POINTS_PER_BLOCK 128
#define THREADS          256
#define FEAT_STRIDE      40   // 32 feats + pad; 80B rows -> 16B aligned, conflict-free

// scale_l = 16 * (256^(1/15))^l - 1 ; res_l = ceil(scale_l)+1  (precomputed)
__constant__ float kScale[NUM_LEVELS] = {
    15.0f,        22.156309f,   32.513413f,   47.502939f,
    69.196818f,   100.593690f,  146.033426f,  211.796914f,
    307.974580f,  444.722150f,  644.080003f,  932.604546f,
    1350.177115f, 1954.517066f, 2829.159829f, 4095.0f
};
__constant__ uint32_t kRes[NUM_LEVELS] = {
    16u, 24u, 34u, 49u, 71u, 102u, 148u, 213u,
    309u, 446u, 646u, 934u, 1352u, 1956u, 2831u, 4096u
};
// res^3 > HASHMAP_SIZE starting at level 5

// ---------------- Fragment helpers ----------------

// A-matrix (16x32 f16), contiguous K order:
//   halves 0..15 : K = (lane<16 ? 0 : 16) + 0..15
__device__ __forceinline__ v16h load_a_frag(const _Float16* rowPtr, int k0) {
    v8h lo = *(const v8h*)(rowPtr + k0);
    v8h hi = *(const v8h*)(rowPtr + k0 + 8);
    v16h a;
#pragma unroll
    for (int i = 0; i < 8; ++i) { a[i] = lo[i]; a[8 + i] = hi[i]; }
    return a;
}

// B-matrix (32x16 f16): lane holds column j = lane&15, halves = K
//   lanes 0-15: K=0..15, lanes 16-31: K=16..31.  B[k][j] = W[j][k] (h = x @ W^T)
__device__ __forceinline__ v16h load_w_frag(const float* __restrict__ w) {
    v16h r;
#pragma unroll
    for (int q = 0; q < 4; ++q) {
        float4 f = *(const float4*)(w + 4 * q);
        r[4 * q + 0] = (_Float16)f.x;
        r[4 * q + 1] = (_Float16)f.y;
        r[4 * q + 2] = (_Float16)f.z;
        r[4 * q + 3] = (_Float16)f.w;
    }
    return r;
}

// ---------------- Kernel ----------------
__global__ __launch_bounds__(THREADS, 2)
void mlptex3d_kernel(const float* __restrict__ texc,
                     const float* __restrict__ aabb,    // [2][3]
                     const float* __restrict__ minmax,  // [2][9]
                     const float* __restrict__ emb,     // [16][524288][2]
                     const float* __restrict__ W0,      // [32][32]
                     const float* __restrict__ W1,      // [32][32]
                     const float* __restrict__ W2,      // [9][32]
                     float* __restrict__ out,           // [N][9]
                     int npts)
{
    __shared__ __align__(16) _Float16 sFeats[POINTS_PER_BLOCK][FEAT_STRIDE];

    const int tid       = threadIdx.x;
    const int blockBase = blockIdx.x * POINTS_PER_BLOCK;

    // ================= Phase 1: hash-grid encode =================
    // 2 threads per point; thread handles 8 levels (64 float2 gathers).
    {
        const int pl   = tid >> 1;
        const int half = tid & 1;
        const int p    = blockBase + pl;

        if (p < npts) {
            // AABB normalize + clamp (scalar-uniform loads)
            const float a0x = aabb[0], a0y = aabb[1], a0z = aabb[2];
            const float r0  = 1.0f / (aabb[3] - a0x);
            const float r1  = 1.0f / (aabb[4] - a0y);
            const float r2  = 1.0f / (aabb[5] - a0z);
            float x = fminf(fmaxf((texc[p * 3 + 0] - a0x) * r0, 0.0f), 1.0f);
            float y = fminf(fmaxf((texc[p * 3 + 1] - a0y) * r1, 0.0f), 1.0f);
            float z = fminf(fmaxf((texc[p * 3 + 2] - a0z) * r2, 0.0f), 1.0f);

#pragma unroll
            for (int i = 0; i < 8; ++i) {
                const int l        = half * 8 + i;
                const float scale  = kScale[l];
                const uint32_t res = kRes[l];
                const bool hashed  = (l >= 5);

                const float px = x * scale + 0.5f;
                const float py = y * scale + 0.5f;
                const float pz = z * scale + 0.5f;
                const float fx0 = floorf(px), fy0 = floorf(py), fz0 = floorf(pz);
                const float fx = px - fx0, fy = py - fy0, fz = pz - fz0;
                const uint32_t ix = (uint32_t)fx0;
                const uint32_t iy = (uint32_t)fy0;
                const uint32_t iz = (uint32_t)fz0;

                const float* tab = emb + (size_t)l * (HASHMAP_SIZE * LEVEL_DIM);
                float acc0 = 0.0f, acc1 = 0.0f;
#pragma unroll
                for (int c = 0; c < 8; ++c) {
                    const uint32_t cx = ix + (uint32_t)(c & 1);
                    const uint32_t cy = iy + (uint32_t)((c >> 1) & 1);
                    const uint32_t cz = iz + (uint32_t)((c >> 2) & 1);
                    uint32_t idx;
                    if (hashed) {
                        idx = (cx ^ (cy * 2654435761u) ^ (cz * 805459861u))
                              & (HASHMAP_SIZE - 1u);
                    } else {
                        idx = cx + cy * res + cz * res * res;
                    }
                    const float wx = (c & 1) ? fx : 1.0f - fx;
                    const float wy = (c & 2) ? fy : 1.0f - fy;
                    const float wz = (c & 4) ? fz : 1.0f - fz;
                    const float w  = wx * wy * wz;
                    const float2 e = *(const float2*)(tab + (size_t)idx * LEVEL_DIM);
                    acc0 = fmaf(w, e.x, acc0);
                    acc1 = fmaf(w, e.y, acc1);
                }
                sFeats[pl][16 * half + 2 * i + 0] = (_Float16)acc0;
                sFeats[pl][16 * half + 2 * i + 1] = (_Float16)acc1;
            }
        }
    }

    __syncthreads();

    // ================= Phase 2: MLP via WMMA =================
    const int  lane    = tid & 31;
    const int  wv      = tid >> 5;
    const int  rowBase = wv * 16;          // this wave owns 16 LDS rows / 16 points
    const int  mrow    = lane & 15;
    const bool hiHalf  = (lane >= 16);
    const int  k0      = hiHalf ? 16 : 0;  // A-fragment K origin (halves 0..15)
    const int  j       = lane & 15;        // B column within 16-wide tile
    const int  kb      = hiHalf ? 16 : 0;  // B-fragment K origin

    // Weight fragments (registers, built once; L2-resident loads)
    const v16h b0a = load_w_frag(W0 + (size_t)(j)      * HID + kb);
    const v16h b0b = load_w_frag(W0 + (size_t)(j + 16) * HID + kb);
    const v16h b1a = load_w_frag(W1 + (size_t)(j)      * HID + kb);
    const v16h b1b = load_w_frag(W1 + (size_t)(j + 16) * HID + kb);
    v16h b2 = {};
    if (j < CHANNELS) b2 = load_w_frag(W2 + (size_t)j * HID + kb);

    const _Float16* myRow = &sFeats[rowBase + mrow][0];
    const v8f zeroC = {};

    // D-fragment -> LDS f16 (with ReLU); lane writes (M = r + mb, N = j / j+16)
    const int mb = hiHalf ? 8 : 0;

    // ---- layer 0 ----
    v16h a = load_a_frag(myRow, k0);
    v8f d0a = __builtin_amdgcn_wmma_f32_16x16x32_f16(false, a, false, b0a,
                                                     (short)0, zeroC, false, false);
    v8f d0b = __builtin_amdgcn_wmma_f32_16x16x32_f16(false, a, false, b0b,
                                                     (short)0, zeroC, false, false);
#pragma unroll
    for (int r = 0; r < 8; ++r) {
        _Float16* dst = &sFeats[rowBase + mb + r][0];
        dst[j]      = (_Float16)fmaxf(d0a[r], 0.0f);
        dst[j + 16] = (_Float16)fmaxf(d0b[r], 0.0f);
    }

    // ---- layer 1 ---- (same-wave LDS store->load is in-order; no barrier)
    a = load_a_frag(myRow, k0);
    v8f d1a = __builtin_amdgcn_wmma_f32_16x16x32_f16(false, a, false, b1a,
                                                     (short)0, zeroC, false, false);
    v8f d1b = __builtin_amdgcn_wmma_f32_16x16x32_f16(false, a, false, b1b,
                                                     (short)0, zeroC, false, false);
#pragma unroll
    for (int r = 0; r < 8; ++r) {
        _Float16* dst = &sFeats[rowBase + mb + r][0];
        dst[j]      = (_Float16)fmaxf(d1a[r], 0.0f);
        dst[j + 16] = (_Float16)fmaxf(d1b[r], 0.0f);
    }

    // ---- layer 2 (9 outputs in one 16-wide tile) ----
    a = load_a_frag(myRow, k0);
    v8f d2 = __builtin_amdgcn_wmma_f32_16x16x32_f16(false, a, false, b2,
                                                    (short)0, zeroC, false, false);

    // ---- sigmoid * (mm1-mm0) + mm0, predicated store of 9 valid columns ----
    const bool colOK = (j < CHANNELS);
    const float mm0  = colOK ? minmax[j]            : 0.0f;
    const float mm1  = colOK ? minmax[CHANNELS + j] : 0.0f;
    const float mmd  = mm1 - mm0;
#pragma unroll
    for (int r = 0; r < 8; ++r) {
        const int p = blockBase + rowBase + mb + r;
        if (colOK && p < npts) {
            const float sg = 1.0f / (1.0f + __expf(-d2[r]));
            out[(size_t)p * CHANNELS + j] = fmaf(sg, mmd, mm0);
        }
    }
}

// ---------------- Launch ----------------
extern "C" void kernel_launch(void* const* d_in, const int* in_sizes, int n_in,
                              void* d_out, int out_size, void* d_ws, size_t ws_size,
                              hipStream_t stream) {
    const float* texc   = (const float*)d_in[0];
    const float* aabb   = (const float*)d_in[1];
    const float* minmax = (const float*)d_in[2];
    const float* emb    = (const float*)d_in[3];
    const float* W0     = (const float*)d_in[4];
    const float* W1     = (const float*)d_in[5];
    const float* W2     = (const float*)d_in[6];
    float* out          = (float*)d_out;

    const int npts   = in_sizes[0] / 3;
    const int blocks = (npts + POINTS_PER_BLOCK - 1) / POINTS_PER_BLOCK;

    hipLaunchKernelGGL(mlptex3d_kernel, dim3(blocks), dim3(THREADS), 0, stream,
                       texc, aabb, minmax, emb, W0, W1, W2, out, npts);
}
